// MaskVectorQuantize_2911987826785
// MI455X (gfx1250) — matrix operand order repaired and checked by reading, not compiled, not run on hardware
//
#include <hip/hip_runtime.h>
#include <float.h>

typedef __attribute__((ext_vector_type(16))) __bf16 v16bf;
typedef __attribute__((ext_vector_type(8)))  float  v8f;
typedef __attribute__((ext_vector_type(4)))  unsigned int v4u;
typedef __attribute__((ext_vector_type(8)))  int v8i;
typedef __attribute__((ext_vector_type(4)))  int v4i;

#define N_EMBED   8192
#define CDIM      256
#define NROWS     16384      // B*H*W
#define HW        1024       // H*W
#define XQ_ELEMS  4194304    // B*C*H*W
#define ROWS_WG   64
#define NCHUNK    64
#define NCHTOT    (N_EMBED / NCHUNK)
#define LDSS      264        // padded LDS row stride (ushorts): 512B data + 16B pad

#if __has_builtin(__builtin_amdgcn_tensor_load_to_lds) && __has_builtin(__builtin_amdgcn_s_wait_tensorcnt)
#define USE_TDM 1
#define NBUF 2
#else
#define USE_TDM 0
#define NBUF 1
#endif

union FragBF { v16bf v; uint4 q[2]; };

#define WMMA_BF16(A, B, C) \
  __builtin_amdgcn_wmma_f32_16x16x32_bf16(false, (A), false, (B), (short)0, (C), false, false)

__device__ __forceinline__ unsigned short f32_to_bf16(float f) {
  unsigned int u = __float_as_uint(f);
  u += 0x7FFFu + ((u >> 16) & 1u);            // round-to-nearest-even
  return (unsigned short)(u >> 16);
}
__device__ __forceinline__ float bf16_to_f32(unsigned short h) {
  return __uint_as_float(((unsigned int)h) << 16);
}

#if USE_TDM
// TDM 2D tile load: 64 rows x 512B (bf16 codebook chunk plane), dword elements,
// hardware-inserted 16B LDS pad per 512B row (-> 528B rows == LDSS ushorts).
__device__ __forceinline__ void tdm_tile_load(unsigned int lds_off,
                                              const unsigned short* gsrc) {
  unsigned long long ga = (unsigned long long)gsrc;
  v4u g0;
  g0[0] = 1u;                                               // count=1 (user descriptor)
  g0[1] = lds_off;                                          // lds_addr (bytes)
  g0[2] = (unsigned int)ga;                                 // global_addr[31:0]
  g0[3] = (unsigned int)((ga >> 32) & 0x01FFFFFFull)        // global_addr[56:32]
          | (2u << 30);                                     // type=2 ("image")
  v8i g1;
  g1[0] = (int)((2u << 16)    // data_size=4B
              | (1u << 20)    // pad_enable
              | (6u << 22)    // pad_interval code 6 -> every 128 DWORDs (512B)
              | (3u << 25));  // pad_amount code 3 -> 4 DWORDs (16B)
  g1[1] = (int)(128u << 16);  // tensor_dim0[15:0]=128 dwords
  g1[2] = (int)(8192u << 16); // tensor_dim0 hi=0 | tensor_dim1[15:0]=8192 rows
  g1[3] = (int)(128u << 16);  // tensor_dim1 hi=0 | tile_dim0=128 dwords
  g1[4] = 64;                 // tile_dim1=64 rows | tile_dim2=0
  g1[5] = 128;                // tensor_dim0_stride[31:0]=128 dwords
  g1[6] = 0;                  // stride hi | tensor_dim1_stride lo
  g1[7] = 0;
  v4i z4 = {0, 0, 0, 0};
  v8i z8 = {0, 0, 0, 0, 0, 0, 0, 0};
  // 6-arg toolchain form: (g0, g1, g2, g3, extra, cpol)
  __builtin_amdgcn_tensor_load_to_lds(g0, g1, z4, z4, z8, 0);
}
#endif

// ---------------- K0: codebook -> bf16 hi/lo planes + c_sq ----------------
__global__ void vq_prep_kernel(const float* __restrict__ w,
                               unsigned short* __restrict__ cb_hi,
                               unsigned short* __restrict__ cb_lo,
                               float* __restrict__ c_sq) {
  __shared__ float rs[256];
  const int n = blockIdx.x;
  const int k = threadIdx.x;
  float v = w[n * CDIM + k];
  unsigned short h = f32_to_bf16(v);
  cb_hi[n * CDIM + k] = h;
  cb_lo[n * CDIM + k] = f32_to_bf16(v - bf16_to_f32(h));
  rs[k] = v * v;
  __syncthreads();
  for (int s = 128; s > 0; s >>= 1) {
    if (k < s) rs[k] += rs[k + s];
    __syncthreads();
  }
  if (k == 0) c_sq[n] = rs[0];
}

// ---------------- K1: distances via bf16x3 WMMA + running argmin ----------------
__global__ __launch_bounds__(256, 1)
void vq_argmin_kernel(const float* __restrict__ x,
                      const unsigned short* __restrict__ cb_hi,
                      const unsigned short* __restrict__ cb_lo,
                      const float* __restrict__ c_sq,
                      int* __restrict__ codes) {
  __shared__ __align__(16) unsigned short sA[2][ROWS_WG][LDSS];        // x tile hi/lo
  __shared__ __align__(16) unsigned short sB[NBUF][2][NCHUNK][LDSS];   // cb chunk hi/lo
  __shared__ float sCsq[NCHUNK];
  __shared__ float red_d[ROWS_WG * 2];
  __shared__ int   red_i[ROWS_WG * 2];

  const int tid    = threadIdx.x;
  const int lane   = tid & 31;
  const int wid    = tid >> 5;
  const int half   = lane >> 4;
  const int l16    = lane & 15;
  const int t      = wid >> 1;   // 16-row tile (0..3), shared by a wave pair
  const int parity = wid & 1;    // wave pair split: parity 0 -> n-tiles {0,1}, 1 -> {2,3}

  const int rowbase = blockIdx.x * ROWS_WG;
  const int b   = rowbase >> 10;     // 64 rows never straddle a batch image
  const int hw0 = rowbase & 1023;

  // Stage x tile (64 rows x 256 ch), split fp32 -> bf16 hi + bf16 lo. Coalesced in hw.
  for (int i = tid; i < ROWS_WG * CDIM; i += 256) {
    int c = i >> 6, dr = i & 63;
    float v = x[(b * CDIM + c) * HW + hw0 + dr];
    unsigned short h = f32_to_bf16(v);
    sA[0][dr][c] = h;
    sA[1][dr][c] = f32_to_bf16(v - bf16_to_f32(h));
  }

  float best[8];
  int   bidx[8];
#pragma unroll
  for (int v = 0; v < 8; ++v) { best[v] = FLT_MAX; bidx[v] = 0; }

  const int arow = t * 16 + l16;     // A-fragment row for this lane (M = lane%16)

#if USE_TDM
  // Prologue: wave 0 DMAs chunk 0 into buffer 0 (hi + lo planes).
  if (wid == 0) {
    tdm_tile_load((unsigned int)(unsigned long long)&sB[0][0][0][0], cb_hi);
    tdm_tile_load((unsigned int)(unsigned long long)&sB[0][1][0][0], cb_lo);
  }
#endif

  for (int chunk = 0; chunk < NCHTOT; ++chunk) {
    const int nbase = chunk * NCHUNK;
    const int buf = chunk & (NBUF - 1);
    __syncthreads();   // all waves done with buf^1 (and previous sCsq)
#if USE_TDM
    if (wid == 0) {
      if (chunk + 1 < NCHTOT) {
        // Prefetch next chunk into the other buffer; then ensure current landed.
        const unsigned short* hsrc = cb_hi + (nbase + NCHUNK) * CDIM;
        const unsigned short* lsrc = cb_lo + (nbase + NCHUNK) * CDIM;
        tdm_tile_load((unsigned int)(unsigned long long)&sB[buf ^ 1][0][0][0], hsrc);
        tdm_tile_load((unsigned int)(unsigned long long)&sB[buf ^ 1][1][0][0], lsrc);
        __builtin_amdgcn_s_wait_tensorcnt(2);   // in-order: current chunk's 2 DMAs done
      } else {
        __builtin_amdgcn_s_wait_tensorcnt(0);
      }
    }
    if (tid < NCHUNK) sCsq[tid] = c_sq[nbase + tid];
#else
    for (int i = tid; i < (NCHUNK * CDIM) / 8; i += 256) {
      int n = i >> 5, kk = (i & 31) << 3;
      *(uint4*)&sB[0][0][n][kk] = *(const uint4*)&cb_hi[(nbase + n) * CDIM + kk];
      *(uint4*)&sB[0][1][n][kk] = *(const uint4*)&cb_lo[(nbase + n) * CDIM + kk];
    }
    if (tid < NCHUNK) sCsq[tid] = c_sq[nbase + tid];
#endif
    __syncthreads();

    // Two N-tiles per wave, 4 independent accumulator chains (main + correction each).
    const int nl0 = parity * 32 + l16;
    const int nl1 = nl0 + 16;
    v8f am0 = {}, ac0 = {}, am1 = {}, ac1 = {};
#pragma unroll
    for (int kt = 0; kt < CDIM / 32; ++kt) {
      FragBF ah, al, b0h, b0l, b1h, b1l;
      const int ka = kt * 32 + half * 16;
      ah.q[0] = *(const uint4*)&sA[0][arow][ka];
      ah.q[1] = *(const uint4*)&sA[0][arow][ka + 8];
      al.q[0] = *(const uint4*)&sA[1][arow][ka];
      al.q[1] = *(const uint4*)&sA[1][arow][ka + 8];
      const int kb = kt * 32 + half * 16;
      b0h.q[0] = *(const uint4*)&sB[buf][0][nl0][kb];
      b0h.q[1] = *(const uint4*)&sB[buf][0][nl0][kb + 8];
      b1h.q[0] = *(const uint4*)&sB[buf][0][nl1][kb];
      b1h.q[1] = *(const uint4*)&sB[buf][0][nl1][kb + 8];
      b0l.q[0] = *(const uint4*)&sB[buf][1][nl0][kb];
      b0l.q[1] = *(const uint4*)&sB[buf][1][nl0][kb + 8];
      b1l.q[0] = *(const uint4*)&sB[buf][1][nl1][kb];
      b1l.q[1] = *(const uint4*)&sB[buf][1][nl1][kb + 8];
      // x.c ~= hi.hi + (lo.hi + hi.lo); 4 independent WMMA chains for ILP
      am0 = WMMA_BF16(ah.v, b0h.v, am0);
      am1 = WMMA_BF16(ah.v, b1h.v, am1);
      ac0 = WMMA_BF16(al.v, b0h.v, ac0);
      ac1 = WMMA_BF16(al.v, b1h.v, ac1);
      ac0 = WMMA_BF16(ah.v, b0l.v, ac0);
      ac1 = WMMA_BF16(ah.v, b1l.v, ac1);
    }
    // score = |c|^2 - 2 x.c  (x_sq constant per row -> dropped for argmin)
    const float cs0 = sCsq[nl0], cs1 = sCsq[nl1];
    const int gi0 = nbase + nl0, gi1 = nbase + nl1;
#pragma unroll
    for (int v = 0; v < 8; ++v) {
      float s0 = cs0 - 2.0f * (am0[v] + ac0[v]);
      float s1 = cs1 - 2.0f * (am1[v] + ac1[v]);
      if (s0 < best[v]) { best[v] = s0; bidx[v] = gi0; }   // strict < keeps lowest idx
      if (s1 < best[v]) { best[v] = s1; bidx[v] = gi1; }
    }
  }

  // Reduce over the 16 lanes of each half (each half holds an independent M set).
#pragma unroll
  for (int v = 0; v < 8; ++v) {
#pragma unroll
    for (int off = 8; off >= 1; off >>= 1) {
      float od = __shfl_xor(best[v], off, 32);
      int   oi = __shfl_xor(bidx[v], off, 32);
      if (od < best[v] || (od == best[v] && oi < bidx[v])) { best[v] = od; bidx[v] = oi; }
    }
  }
  // Leaders (lane 0 -> M=v, lane 16 -> M=8+v) publish; then merge the wave pair.
  if (l16 == 0) {
#pragma unroll
    for (int v = 0; v < 8; ++v) {
      int lr = t * 16 + half * 8 + v;
      red_d[lr * 2 + parity] = best[v];
      red_i[lr * 2 + parity] = bidx[v];
    }
  }
  __syncthreads();
  if (tid < ROWS_WG) {
    float d0 = red_d[tid * 2], d1 = red_d[tid * 2 + 1];
    int   i0 = red_i[tid * 2], i1 = red_i[tid * 2 + 1];
    codes[rowbase + tid] = (d1 < d0 || (d1 == d0 && i1 < i0)) ? i1 : i0;
  }
}

// ---------------- K2: exact fp32 gather + transpose-out + per-row SSE ----------------
__global__ void vq_gather_kernel(const float* __restrict__ x,
                                 const float* __restrict__ w,
                                 const int* __restrict__ codes,
                                 float* __restrict__ out,
                                 float* __restrict__ row_sum) {
  __shared__ float rs[256];
  const int r = blockIdx.x;
  const int c = threadIdx.x;
  const int idx = codes[r];
  const int b = r >> 10, hw = r & 1023;
  float xv = x[(b * CDIM + c) * HW + hw];
  float q  = w[idx * CDIM + c];
  out[(b * CDIM + c) * HW + hw] = q;      // x_q = xt + sg(x_q - xt) == x_q numerically
  float d = q - xv;
  rs[c] = d * d;
  __syncthreads();
  for (int s = 128; s > 0; s >>= 1) {
    if (c < s) rs[c] += rs[c + s];
    __syncthreads();
  }
  if (c == 0) {
    row_sum[r] = rs[0];
    out[XQ_ELEMS + 1 + r] = (float)idx;   // code output (float buffer)
  }
}

// ---------------- K3: deterministic loss reduction ----------------
__global__ void vq_loss_kernel(const float* __restrict__ row_sum,
                               float* __restrict__ out) {
  __shared__ float rs[256];
  float s = 0.f;
  for (int i = threadIdx.x; i < NROWS; i += 256) s += row_sum[i];
  rs[threadIdx.x] = s;
  __syncthreads();
  for (int st = 128; st > 0; st >>= 1) {
    if (threadIdx.x < st) rs[threadIdx.x] += rs[threadIdx.x + st];
    __syncthreads();
  }
  if (threadIdx.x == 0)
    out[XQ_ELEMS] = 1.25f * rs[0] / (float)XQ_ELEMS;  // (1 + BETA) * mean
}

extern "C" void kernel_launch(void* const* d_in, const int* in_sizes, int n_in,
                              void* d_out, int out_size, void* d_ws, size_t ws_size,
                              hipStream_t stream) {
  const float* x = (const float*)d_in[0];
  const float* w = (const float*)d_in[1];
  float* out = (float*)d_out;
  char* ws = (char*)d_ws;
  // workspace layout (~8.2 MB total)
  unsigned short* cb_hi   = (unsigned short*)(ws);
  unsigned short* cb_lo   = (unsigned short*)(ws + 4194304);
  float*          c_sq    = (float*)(ws + 8388608);
  int*            codes   = (int*)(ws + 8421376);
  float*          row_sum = (float*)(ws + 8486912);

  vq_prep_kernel<<<N_EMBED, 256, 0, stream>>>(w, cb_hi, cb_lo, c_sq);
  vq_argmin_kernel<<<NROWS / ROWS_WG, 256, 0, stream>>>(x, cb_hi, cb_lo, c_sq, codes);
  vq_gather_kernel<<<NROWS, 256, 0, stream>>>(x, w, codes, out, row_sum);
  vq_loss_kernel<<<1, 256, 0, stream>>>(row_sum, out);
}
